// CausalSelfAttention_644245095083
// MI455X (gfx1250) — matrix operand order repaired and checked by reading, not compiled, not run on hardware
//
#include <hip/hip_runtime.h>
#include <hip/hip_bf16.h>

#define NH    16
#define KVHN  4
#define HD    128
#define BATCH 2
#define SEQ   2048
#define DIM   2048
#define KVD   512

typedef __attribute__((ext_vector_type(16))) __bf16 v16bf;
typedef __attribute__((ext_vector_type(8)))  __bf16 v8bf;
typedef __attribute__((ext_vector_type(8)))  float  v8f;
typedef __attribute__((ext_vector_type(4)))  unsigned int v4u;
typedef __attribute__((ext_vector_type(4)))  int v4i;
typedef __attribute__((ext_vector_type(8)))  int v8i;
typedef unsigned short u16;
typedef unsigned int   u32;

__device__ __forceinline__ u16 f2bf(float f){
  u32 u = __float_as_uint(f);
  u = (u + 0x7fffu + ((u >> 16) & 1u)) >> 16;   // round-to-nearest-even
  return (u16)u;
}
__device__ __forceinline__ float bf2f(u16 h){
  return __uint_as_float(((u32)h) << 16);
}
__device__ __forceinline__ v8f vzero(){
  v8f z;
#pragma unroll
  for (int i = 0; i < 8; ++i) z[i] = 0.0f;
  return z;
}
// A fragment (16x32 bf16): lane holds row m = lane&15; k = base + {0..7} and base + {16..23},
// with base already including the (lane>=16 ? 8 : 0) half-wave shift.
__device__ __forceinline__ v16bf fragA(const u16* p){
  v8bf a = *(const v8bf*)p;
  v8bf b = *(const v8bf*)(p + 16);
  v16bf r;
#pragma unroll
  for (int i = 0; i < 8; ++i){ r[i] = a[i]; r[i + 8] = b[i]; }
  return r;
}
// B fragment (32x16 bf16): lane holds column n = lane&15; k = base + {0..7} and base + {16..23},
// with base already including the (lane>=16 ? 8 : 0) half-wave shift.
__device__ __forceinline__ v16bf fragB(const u16* p){
  v8bf a = *(const v8bf*)p;
  v8bf b = *(const v8bf*)(p + 16);
  v16bf r;
#pragma unroll
  for (int i = 0; i < 8; ++i){ r[i] = a[i]; r[i + 8] = b[i]; }
  return r;
}
__device__ __forceinline__ v8f wmma_bf16(v16bf a, v16bf b, v8f c){
  return __builtin_amdgcn_wmma_f32_16x16x32_bf16(false, a, false, b, (short)0, c, false, false);
}

// ---- Tensor Data Mover: 2-D tile (16-bit elements) global -> LDS --------------------
// D# group0: [1:0]=count=1, [63:32]=lds_addr, [120:64]=global_addr, [127:126]=type(2)
// D# group1: [17:16]=data_size(1 => 2B), tensor_dim0/1, tile_dim0/1, tensor_dim0_stride
__device__ __forceinline__ void tdm_load_2d(u32 lds_off, const void* gptr,
                                            u32 tile_d0, u32 tile_d1,
                                            u32 tensor_d0, u32 tensor_d1,
                                            u32 stride0 /* elements */){
  unsigned long long va = (unsigned long long)(uintptr_t)gptr;
  v4u g0;
  g0[0] = 1u;                                   // count=1, user mode
  g0[1] = lds_off;                              // LDS byte address
  g0[2] = (u32)va;                              // global addr [95:64]
  g0[3] = (u32)((va >> 32) & 0x01FFFFFFu) | 0x80000000u;  // addr[56:32] | type=2
  v8i g1;
  g1[0] = (int)(1u << 16);                      // workgroup_mask=0, data_size=1 (2 bytes)
  g1[1] = (int)((tensor_d0 & 0xFFFFu) << 16);   // tensor_dim0 low16 @ bits 63:48
  g1[2] = (int)((tensor_d0 >> 16) | ((tensor_d1 & 0xFFFFu) << 16));
  g1[3] = (int)((tensor_d1 >> 16) | (tile_d0 << 16));      // tile_dim0 @ 127:112
  g1[4] = (int)tile_d1;                         // tile_dim1 @ 143:128, tile_dim2=0
  g1[5] = (int)stride0;                         // tensor_dim0_stride low32
  g1[6] = 0;                                    // stride0 hi16 | stride1 lo16
  g1[7] = 0;
  v4i z4 = {0, 0, 0, 0};
#if __clang_major__ >= 23
  v8i z8 = {0, 0, 0, 0, 0, 0, 0, 0};
  __builtin_amdgcn_tensor_load_to_lds(g0, g1, z4, z4, z8, 0);
#else
  __builtin_amdgcn_tensor_load_to_lds(g0, g1, z4, z4, 0);
#endif
}

// ---------------- elementwise converts ----------------
__global__ void k_f32_to_bf16(const float* __restrict__ in, u16* __restrict__ out, int n){
  int i = blockIdx.x * 256 + threadIdx.x;
  if (i < n) out[i] = f2bf(in[i]);
}

// ---------------- GEMM: C[M,N] = A[M,K] * W[N,K]^T, bf16 in, f32 accum ----------------
template<int STORE_BF16>
__global__ __launch_bounds__(256) void k_gemm(const u16* __restrict__ A,
                                              const u16* __restrict__ W,
                                              void* __restrict__ Cout,
                                              int M, int N, int K){
  const int lane = threadIdx.x & 31;
  const int wave = threadIdx.x >> 5;
  const int wm = wave & 3;            // 4 waves along M
  const int wn = wave >> 2;           // 2 waves along N
  const int nb = N >> 7;
  const int bm = blockIdx.x / nb;
  const int bn = blockIdx.x % nb;
  const int mBase = bm * 128 + wm * 32;
  const int nBase = bn * 128 + wn * 64;
  const int l15  = lane & 15;
  const int hi8  = (lane >> 4) << 3;
  const int hi16 = (lane >> 4) << 4;

  const u16* a0p = A + (size_t)(mBase + l15) * K + hi8;
  const u16* a1p = a0p + (size_t)16 * K;
  const u16* bp0 = W + (size_t)(nBase +  0 + l15) * K + hi8;
  const u16* bp1 = W + (size_t)(nBase + 16 + l15) * K + hi8;
  const u16* bp2 = W + (size_t)(nBase + 32 + l15) * K + hi8;
  const u16* bp3 = W + (size_t)(nBase + 48 + l15) * K + hi8;

  v8f acc[2][4];
#pragma unroll
  for (int i = 0; i < 2; ++i)
#pragma unroll
    for (int j = 0; j < 4; ++j) acc[i][j] = vzero();

  for (int k = 0; k < K; k += 32){
    __builtin_prefetch(a0p + k + 512, 0, 0);
    __builtin_prefetch(bp0 + k + 512, 0, 0);
    v16bf a0 = fragA(a0p + k);
    v16bf a1 = fragA(a1p + k);
    v16bf b0 = fragB(bp0 + k);
    v16bf b1 = fragB(bp1 + k);
    v16bf b2 = fragB(bp2 + k);
    v16bf b3 = fragB(bp3 + k);
    acc[0][0] = wmma_bf16(a0, b0, acc[0][0]);
    acc[1][0] = wmma_bf16(a1, b0, acc[1][0]);
    acc[0][1] = wmma_bf16(a0, b1, acc[0][1]);
    acc[1][1] = wmma_bf16(a1, b1, acc[1][1]);
    acc[0][2] = wmma_bf16(a0, b2, acc[0][2]);
    acc[1][2] = wmma_bf16(a1, b2, acc[1][2]);
    acc[0][3] = wmma_bf16(a0, b3, acc[0][3]);
    acc[1][3] = wmma_bf16(a1, b3, acc[1][3]);
  }

#pragma unroll
  for (int mt = 0; mt < 2; ++mt){
#pragma unroll
    for (int nt = 0; nt < 4; ++nt){
#pragma unroll
      for (int j = 0; j < 8; ++j){
        int row = mBase + mt * 16 + hi8 + j;   // C layout: rows j / j+8 per lane-half
        int col = nBase + nt * 16 + l15;
        if (STORE_BF16) ((u16*)Cout)[(size_t)row * N + col] = f2bf(acc[mt][nt][j]);
        else            ((float*)Cout)[(size_t)row * N + col] = acc[mt][nt][j];
      }
    }
  }
}

// ---------------- RoPE + head-major relayout (bf16 -> bf16) ----------------
__global__ void k_rope(const u16* __restrict__ raw,   // [B*L, Hh*HD]
                       u16* __restrict__ out,         // [B, Hh, L, HD]
                       int Hh){
  int t = blockIdx.x * 256 + threadIdx.x;
  int total = BATCH * SEQ * Hh * HD;
  if (t >= total) return;
  int d  = t & (HD - 1);
  int r1 = t >> 7;
  int h  = r1 % Hh;
  int bl = r1 / Hh;
  int l  = bl & (SEQ - 1);
  int b  = bl >> 11;
  const u16* rowp = raw + (size_t)bl * (Hh * HD) + h * HD;
  float x  = bf2f(rowp[d]);
  int  d2  = (d < 64) ? d + 64 : d - 64;
  float xp = bf2f(rowp[d2]);
  int j = d & 63;
  float inv = __expf(-(float)j * (9.210340371976184f / 64.0f)); // 10000^(-j/64)
  float ang = (float)l * inv;
  float s, c;
  __sincosf(ang, &s, &c);
  float v = x * c + ((d < 64) ? -xp : xp) * s;
  out[(((size_t)b * Hh + h) * SEQ + l) * HD + d] = f2bf(v);
}

// ---------------- V relayout to [B, KVH, HD, L] (bf16 copy/transpose) ----------------
__global__ void k_vt(const u16* __restrict__ raw, u16* __restrict__ out){
  int t = blockIdx.x * 256 + threadIdx.x;
  int total = BATCH * SEQ * KVHN * HD;
  if (t >= total) return;
  int d  = t & (HD - 1);
  int r1 = t >> 7;
  int h  = r1 % KVHN;
  int bl = r1 / KVHN;
  int l  = bl & (SEQ - 1);
  int b  = bl >> 11;
  out[(((size_t)b * KVHN + h) * HD + d) * SEQ + l] = raw[(size_t)bl * KVD + h * HD + d];
}

// ---------------- Flash attention (causal, GQA), bf16 WMMA + f32 softmax --------------
// K/V^T tiles double-buffered in LDS via the Tensor Data Mover; DMA for tile i+1
// overlaps WMMA compute on tile i.
__global__ __launch_bounds__(256) void k_attn(const u16* __restrict__ Q,   // [B,NH,L,HD]
                                              const u16* __restrict__ Kc,  // [B,KVH,L,HD]
                                              const u16* __restrict__ Vt,  // [B,KVH,HD,L]
                                              u16* __restrict__ O){        // [B*L, DIM]
  __shared__ u16 sK[2][64][128];   // keys x d, double buffered (16KB each)
  __shared__ u16 sV[2][128][64];   // d x keys, double buffered (16KB each)
  __shared__ u16 sP[8][16][64];    // per-wave P staging
  const int lane = threadIdx.x & 31;
  const int wave = threadIdx.x >> 5;
  const int l15  = lane & 15;
  const int hi8  = (lane >> 4) << 3;
  const int hi16 = (lane >> 4) << 4;

  int bid  = blockIdx.x;
  int qblk = bid & 15;            // SEQ/128 = 16 q-blocks
  int h    = (bid >> 4) & 15;
  int b    = bid >> 8;
  int kvh  = h >> 2;              // rep = NH/KVH = 4
  int qb0  = qblk * 128;
  int qbase = qb0 + wave * 16;
  const size_t bkv = (size_t)(b * KVHN + kvh);
  const u16* kgBase = Kc + (bkv * SEQ) * HD;
  const u16* vgBase = Vt + (bkv * HD) * SEQ;

  // Q fragments, resident all kernel (A layout)
  const u16* qrow = Q + (((size_t)(b * NH + h)) * SEQ + (qbase + l15)) * HD;
  v16bf qf[4];
#pragma unroll
  for (int kk = 0; kk < 4; ++kk) qf[kk] = fragA(qrow + kk * 32 + hi8);

  v8f oacc[8];
#pragma unroll
  for (int dt = 0; dt < 8; ++dt) oacc[dt] = vzero();
  float mrow[8], lrow[8];
#pragma unroll
  for (int j = 0; j < 8; ++j){ mrow[j] = -1.0e30f; lrow[j] = 0.0f; }

  const float scale = 0.08838834764831845f;   // 1/sqrt(128)
  const int nIter = (qb0 + 128) >> 6;         // 64-key blocks, causal upper bound

  // prologue: DMA tile 0 into buffer 0 (wave 0 drives the TDM)
  if (wave == 0){
    tdm_load_2d((u32)(uintptr_t)&sK[0][0][0], kgBase,
                /*tile*/ HD, 64, /*tensor*/ HD, SEQ, /*stride*/ HD);
    tdm_load_2d((u32)(uintptr_t)&sV[0][0][0], vgBase,
                /*tile*/ 64, HD, /*tensor*/ SEQ, KVHN * HD * BATCH, /*stride*/ SEQ);
  }

  for (int it = 0; it < nIter; ++it){
    const int kb  = it << 6;
    const int buf = it & 1;
    if (wave == 0) __builtin_amdgcn_s_wait_tensorcnt(0);
    __syncthreads();    // tile `buf` ready; all waves done reading buffer buf^1
    if (wave == 0 && it + 1 < nIter){
      const int kb2 = kb + 64, b2 = buf ^ 1;
      tdm_load_2d((u32)(uintptr_t)&sK[b2][0][0], kgBase + (size_t)kb2 * HD,
                  HD, 64, HD, SEQ, HD);
      tdm_load_2d((u32)(uintptr_t)&sV[b2][0][0], vgBase + kb2,
                  64, HD, SEQ, KVHN * HD * BATCH, SEQ);
    }

    if (kb <= qbase + 15){
      // S = Q K^T (4 key sub-tiles x 4 d-chunks)
      v8f st[4];
#pragma unroll
      for (int nt = 0; nt < 4; ++nt){
        v8f acc = vzero();
#pragma unroll
        for (int kk = 0; kk < 4; ++kk){
          v16bf bfr = fragB(&sK[buf][nt * 16 + l15][kk * 32 + hi8]);
          acc = wmma_bf16(qf[kk], bfr, acc);
        }
        st[nt] = acc;
      }
      // scale + causal mask (C layout: row = hi8+j, col = nt*16+l15)
#pragma unroll
      for (int nt = 0; nt < 4; ++nt){
        int key = kb + nt * 16 + l15;
#pragma unroll
        for (int j = 0; j < 8; ++j){
          float s = st[nt][j] * scale;
          int q = qbase + hi8 + j;
          st[nt][j] = (key <= q) ? s : -1.0e30f;
        }
      }
      // online softmax (row reductions across 16 lanes within each half-wave)
#pragma unroll
      for (int j = 0; j < 8; ++j){
        float v = fmaxf(fmaxf(st[0][j], st[1][j]), fmaxf(st[2][j], st[3][j]));
        v = fmaxf(v, __shfl_xor(v, 1, 32));
        v = fmaxf(v, __shfl_xor(v, 2, 32));
        v = fmaxf(v, __shfl_xor(v, 4, 32));
        v = fmaxf(v, __shfl_xor(v, 8, 32));
        float mn = fmaxf(mrow[j], v);
        float alpha = __expf(mrow[j] - mn);
        mrow[j] = mn;
        lrow[j] *= alpha;
#pragma unroll
        for (int dt = 0; dt < 8; ++dt) oacc[dt][j] *= alpha;
        float rs = 0.0f;
#pragma unroll
        for (int nt = 0; nt < 4; ++nt){
          float p = __expf(st[nt][j] - mn);
          st[nt][j] = p;
          rs += p;
        }
        rs += __shfl_xor(rs, 1, 32);
        rs += __shfl_xor(rs, 2, 32);
        rs += __shfl_xor(rs, 4, 32);
        rs += __shfl_xor(rs, 8, 32);
        lrow[j] += rs;
      }
      // C-layout -> A-layout via per-wave LDS staging (same-wave DS ops are in order)
#pragma unroll
      for (int nt = 0; nt < 4; ++nt)
#pragma unroll
        for (int j = 0; j < 8; ++j)
          sP[wave][hi8 + j][nt * 16 + l15] = f2bf(st[nt][j]);
      v16bf pf0 = fragA(&sP[wave][l15][ 0 + hi8]);
      v16bf pf1 = fragA(&sP[wave][l15][32 + hi8]);
      // O += P V
#pragma unroll
      for (int dt = 0; dt < 8; ++dt){
        v16bf v0 = fragB(&sV[buf][dt * 16 + l15][ 0 + hi8]);
        v16bf v1 = fragB(&sV[buf][dt * 16 + l15][32 + hi8]);
        oacc[dt] = wmma_bf16(pf0, v0, oacc[dt]);
        oacc[dt] = wmma_bf16(pf1, v1, oacc[dt]);
      }
    }
  }

  // normalize + store into token-major [B*L, DIM] bf16 buffer for the Wo GEMM
#pragma unroll
  for (int j = 0; j < 8; ++j){
    float inv = 1.0f / lrow[j];
    int q = qbase + hi8 + j;
    u16* orow = O + ((size_t)(b * SEQ) + q) * DIM + h * HD;
#pragma unroll
    for (int dt = 0; dt < 8; ++dt)
      orow[dt * 16 + l15] = f2bf(oacc[dt][j] * inv);
  }
}

extern "C" void kernel_launch(void* const* d_in, const int* in_sizes, int n_in,
                              void* d_out, int out_size, void* d_ws, size_t ws_size,
                              hipStream_t stream){
  const float* x  = (const float*)d_in[0];
  const float* Wq = (const float*)d_in[1];
  const float* Wk = (const float*)d_in[2];
  const float* Wv = (const float*)d_in[3];
  const float* Wo = (const float*)d_in[4];

  char* ws = (char*)d_ws;
  size_t off = 0;
  auto alloc = [&](size_t bytes) -> char* {
    char* p = ws + off;
    off += (bytes + 255) & ~(size_t)255;
    return p;
  };
  const size_t M = (size_t)BATCH * SEQ;                 // 4096
  u16* xb   = (u16*)alloc(M * DIM * 2);                 // x  bf16  [M, 2048]
  u16* wqb  = (u16*)alloc((size_t)DIM * DIM * 2);
  u16* wkb  = (u16*)alloc((size_t)KVD * DIM * 2);
  u16* wvb  = (u16*)alloc((size_t)KVD * DIM * 2);
  u16* wob  = (u16*)alloc((size_t)DIM * DIM * 2);
  u16* qraw = (u16*)alloc(M * DIM * 2);                 // [M, 2048]
  u16* kraw = (u16*)alloc(M * KVD * 2);                 // [M, 512]
  u16* vraw = (u16*)alloc(M * KVD * 2);
  u16* qbf  = (u16*)alloc(M * DIM * 2);                 // [B,NH,L,HD]
  u16* kbf  = (u16*)alloc(M * KVD * 2);                 // [B,KVH,L,HD]
  u16* vtbf = (u16*)alloc(M * KVD * 2);                 // [B,KVH,HD,L]
  u16* obf  = xb;   // x-bf16 is dead after the projections; reuse for O [M, 2048]

  // 1) f32 -> bf16 converts
  {
    int n;
    n = (int)(M * DIM);          k_f32_to_bf16<<<(n + 255) / 256, 256, 0, stream>>>(x,  xb,  n);
    n = DIM * DIM;               k_f32_to_bf16<<<(n + 255) / 256, 256, 0, stream>>>(Wq, wqb, n);
    n = KVD * DIM;               k_f32_to_bf16<<<(n + 255) / 256, 256, 0, stream>>>(Wk, wkb, n);
    n = KVD * DIM;               k_f32_to_bf16<<<(n + 255) / 256, 256, 0, stream>>>(Wv, wvb, n);
    n = DIM * DIM;               k_f32_to_bf16<<<(n + 255) / 256, 256, 0, stream>>>(Wo, wob, n);
  }
  // 2) projections: Q/K/V = X W^T  (bf16 WMMA, bf16 epilogue)
  k_gemm<1><<<(4096 / 128) * (DIM / 128), 256, 0, stream>>>(xb, wqb, qraw, 4096, DIM, DIM);
  k_gemm<1><<<(4096 / 128) * (KVD / 128), 256, 0, stream>>>(xb, wkb, kraw, 4096, KVD, DIM);
  k_gemm<1><<<(4096 / 128) * (KVD / 128), 256, 0, stream>>>(xb, wvb, vraw, 4096, KVD, DIM);
  // 3) RoPE + relayout
  {
    int n = (int)(M * DIM);      k_rope<<<(n + 255) / 256, 256, 0, stream>>>(qraw, qbf, NH);
    n = (int)(M * KVD);          k_rope<<<(n + 255) / 256, 256, 0, stream>>>(kraw, kbf, KVHN);
    n = (int)(M * KVD);          k_vt  <<<(n + 255) / 256, 256, 0, stream>>>(vraw, vtbf);
  }
  // 4) causal flash attention: 1 block per (b, h, 128 q-rows)
  k_attn<<<BATCH * NH * (SEQ / 128), 256, 0, stream>>>(qbf, kbf, vtbf, obf);
  // 5) output projection: out = O Wo^T  (f32 epilogue into d_out)
  k_gemm<0><<<(4096 / 128) * (DIM / 128), 256, 0, stream>>>(obf, wob, (float*)d_out, 4096, DIM, DIM);
}
